// MultiHeadSelfAttention_44023414784383
// MI455X (gfx1250) — matrix operand order repaired and checked by reading, not compiled, not run on hardware
//
#include <hip/hip_runtime.h>
#include <hip/hip_bf16.h>

typedef __attribute__((ext_vector_type(16))) _Float16 v16h;
typedef __attribute__((ext_vector_type(8)))  _Float16 v8h;
typedef __attribute__((ext_vector_type(8)))  float    v8f;
typedef __attribute__((ext_vector_type(4)))  unsigned int u32x4;
typedef __attribute__((ext_vector_type(4)))  int          i32x4;
typedef __attribute__((ext_vector_type(8)))  int          i32x8;

union HFrag { v16h v; v8h h[2]; };

#define EMBED 1024
#define HEADS 16
#define DH    64
#define BATCH 2
#define SEQ   2048
#define ROWS  (BATCH*SEQ)   // 4096
#define LDSTR 40            // padded half-stride for 32-wide rows (80B, 16B aligned)
#define KSTR  72            // padded half-stride for 64-wide rows (144B, 16B aligned)

// TDM pad configs: pad_enable<<20 | pad_interval<<22 | pad_amount<<25
// 32-half rows: pad 4 dwords after every 16 dwords  -> stride LDSTR
#define PADF32 ((1u << 20) | (3u << 22) | (3u << 25))
// 64-half rows: pad 4 dwords after every 32 dwords  -> stride KSTR
#define PADF64 ((1u << 20) | (4u << 22) | (3u << 25))

#if defined(__has_builtin)
#if __has_builtin(__builtin_amdgcn_tensor_load_to_lds) && __has_builtin(__builtin_amdgcn_s_wait_tensorcnt)
#define HAVE_TDM 1
#endif
#endif
#if defined(__has_include)
#if __has_include(<hip/amd_detail/amd_gfx1250_TDM.h>)
#define TDM_ARITY6 1
#endif
#endif

#if defined(HAVE_TDM)
// Issue one TDM 2D tile load (global -> LDS), f16 elements.
// D# per CDNA5 ISA ch.8: group0 = {count=1, lds_addr, global_addr(57b), type=2};
// group1 = {data_size=2B | pad cfg, tensor dims, tile dims, dim0 stride}.
__device__ __forceinline__ void tdm_load_2d(const void* gptr, const void* lptr,
                                            unsigned td0, unsigned td1,
                                            unsigned tile0, unsigned tile1,
                                            unsigned long long stride0,
                                            unsigned padflags) {
  unsigned long long ga = (unsigned long long)(uintptr_t)gptr;
  unsigned la = (unsigned)(uintptr_t)lptr;          // low 32 bits of generic ptr = LDS offset
  u32x4 g0 = { 1u,
               la,
               (unsigned)ga,
               (unsigned)(ga >> 32) | (2u << 30) };  // addr[56:32] | type=2
  i32x8 g1 = {
    (int)(0x10000u | padflags),                      // data_size=1 (2 bytes) + pad cfg
    (int)((td0 & 0xFFFFu) << 16),                    // tensor_dim0[15:0]
    (int)((td0 >> 16) | ((td1 & 0xFFFFu) << 16)),    // tensor_dim0[31:16] | tensor_dim1[15:0]
    (int)((td1 >> 16) | (tile0 << 16)),              // tensor_dim1[31:16] | tile_dim0
    (int)(tile1 & 0xFFFFu),                          // tile_dim1 (tile_dim2 = 0)
    (int)(unsigned)(stride0 & 0xFFFFFFFFull),        // tensor_dim0_stride[31:0]
    (int)(unsigned)((stride0 >> 32) & 0xFFFFull),    // stride[47:32] (dim1_stride = 0)
    0 };
  i32x4 z4 = {0, 0, 0, 0};
#if defined(TDM_ARITY6)
  i32x8 z8 = {0, 0, 0, 0, 0, 0, 0, 0};
  __builtin_amdgcn_tensor_load_to_lds(g0, g1, z4, z4, z8, 0);
#else
  __builtin_amdgcn_tensor_load_to_lds(g0, g1, z4, z4, 0);
#endif
}
#endif

// ---------------- f32 -> f16 conversion ----------------
__global__ void cvt_f32_f16(const float* __restrict__ in, _Float16* __restrict__ out, int n) {
  int i = blockIdx.x * blockDim.x + threadIdx.x;
  if (i < n) out[i] = (_Float16)in[i];
}

// ---------------- f32 -> f16 transpose (1024x1024): out[n][k] = in[k][n] ----------------
__global__ __launch_bounds__(256)
void transpose_cvt_f32_f16(const float* __restrict__ in, _Float16* __restrict__ out) {
  __shared__ float tile[32][33];
  const int bx = blockIdx.x * 32;   // input col block
  const int by = blockIdx.y * 32;   // input row block
  const int tx = threadIdx.x & 31;
  const int ty = threadIdx.x >> 5;  // 8 rows per pass
  #pragma unroll
  for (int i = 0; i < 4; ++i) {
    int row = ty + i * 8;
    tile[row][tx] = in[(size_t)(by + row) * EMBED + bx + tx];
  }
  __syncthreads();
  #pragma unroll
  for (int i = 0; i < 4; ++i) {
    int row = ty + i * 8;           // output row (= input col) within block
    out[(size_t)(bx + row) * EMBED + by + tx] = (_Float16)tile[tx][row];
  }
}

// ---------------- tiled WMMA GEMM: C = A(4096x1024) * Wt^T + bias ----------------
// Wt is the PRE-TRANSPOSED weight: Wt[n][k] = W[k][n]  (so both tiles stage row-major)
// mode 0: f16 out, head-split [B,H,S,Dh]      (Q, K)
// mode 1: f16 out, transposed [B,H,Dh,S]      (V)
// mode 2: f32 out, row-major [ROWS,EMBED]     (final projection)
__global__ __launch_bounds__(256)
void gemm_wmma(const _Float16* __restrict__ A, const _Float16* __restrict__ Wt,
               const float* __restrict__ bias, _Float16* __restrict__ outh,
               float* __restrict__ outf, int mode) {
  __shared__ _Float16 As[2][128 * LDSTR];
  __shared__ _Float16 Bs[2][128 * LDSTR];

  const int m0   = blockIdx.y * 128;
  const int n0   = blockIdx.x * 128;
  const int t    = threadIdx.x;
  const int wave = t >> 5;
  const int lane = t & 31;
  const int lm   = lane & 15;
  const int hi   = (lane >> 4) & 1;
  const int wm   = wave >> 2;   // 0..1 : 64-row strip
  const int wn   = wave & 3;    // 0..3 : 32-col strip
  const int NT   = EMBED / 32;

  v8f zero = {0.f,0.f,0.f,0.f,0.f,0.f,0.f,0.f};
  v8f acc[4][2];
  #pragma unroll
  for (int i = 0; i < 4; ++i)
    #pragma unroll
    for (int j = 0; j < 2; ++j) acc[i][j] = zero;

#if defined(HAVE_TDM)
  if (wave == 0) {   // prologue: stage tile 0 into buffer 0
    tdm_load_2d(A  + (size_t)m0 * EMBED, &As[0][0], 32, 128, 32, 128,
                (unsigned long long)EMBED, PADF32);
    tdm_load_2d(Wt + (size_t)n0 * EMBED, &Bs[0][0], 32, 128, 32, 128,
                (unsigned long long)EMBED, PADF32);
  }
#endif

  for (int kt = 0; kt < NT; ++kt) {
    const int cur = kt & 1;
#if defined(HAVE_TDM)
    if (wave == 0) {
      if (kt + 1 < NT) {   // prefetch next tile pair into the other buffer
        tdm_load_2d(A  + (size_t)m0 * EMBED + (size_t)(kt + 1) * 32, &As[cur ^ 1][0],
                    32, 128, 32, 128, (unsigned long long)EMBED, PADF32);
        tdm_load_2d(Wt + (size_t)n0 * EMBED + (size_t)(kt + 1) * 32, &Bs[cur ^ 1][0],
                    32, 128, 32, 128, (unsigned long long)EMBED, PADF32);
        __builtin_amdgcn_s_wait_tensorcnt(2);   // in-order: current pair has landed
      } else {
        __builtin_amdgcn_s_wait_tensorcnt(0);
      }
    }
#else
    // fallback: cooperative staging (both tiles contiguous thanks to Wt)
    {
      int row = t >> 1, seg = (t & 1) * 16;
      const _Float16* ga = A  + (size_t)(m0 + row) * EMBED + kt * 32 + seg;
      const _Float16* gb = Wt + (size_t)(n0 + row) * EMBED + kt * 32 + seg;
      *(v8h*)&As[cur][row * LDSTR + seg]     = *(const v8h*)ga;
      *(v8h*)&As[cur][row * LDSTR + seg + 8] = *(const v8h*)(ga + 8);
      *(v8h*)&Bs[cur][row * LDSTR + seg]     = *(const v8h*)gb;
      *(v8h*)&Bs[cur][row * LDSTR + seg + 8] = *(const v8h*)(gb + 8);
    }
#endif
    __syncthreads();

    HFrag bf[2];
    #pragma unroll
    for (int nj = 0; nj < 2; ++nj) {
      int n = wn * 32 + nj * 16 + lm;
      int base = n * LDSTR + hi * 8;             // lanes<16: K {0..7,16..23}; hi: {8..15,24..31}
      bf[nj].h[0] = *(const v8h*)&Bs[cur][base];
      bf[nj].h[1] = *(const v8h*)&Bs[cur][base + 16];
    }
    #pragma unroll
    for (int mi = 0; mi < 4; ++mi) {
      int r = wm * 64 + mi * 16 + lm;
      int base = r * LDSTR + hi * 8;             // lanes<16: K {0..7,16..23}; hi: {8..15,24..31}
      HFrag af;
      af.h[0] = *(const v8h*)&As[cur][base];
      af.h[1] = *(const v8h*)&As[cur][base + 16];
      #pragma unroll
      for (int nj = 0; nj < 2; ++nj)
        acc[mi][nj] = __builtin_amdgcn_wmma_f32_16x16x32_f16(
            false, af.v, false, bf[nj].v, (short)0, acc[mi][nj], false, false);
    }
    __syncthreads();   // readers done before this buffer is DMA-overwritten again
  }

  // epilogue
  #pragma unroll
  for (int mi = 0; mi < 4; ++mi) {
    #pragma unroll
    for (int nj = 0; nj < 2; ++nj) {
      int gn = n0 + wn * 32 + nj * 16 + lm;
      float bv = bias[gn];
      #pragma unroll
      for (int r = 0; r < 8; ++r) {
        int gm = m0 + wm * 64 + mi * 16 + hi * 8 + r;   // C layout: VGPR r -> M = r + 8*hi
        float val = acc[mi][nj][r] + bv;
        if (mode == 2) {
          outf[(size_t)gm * EMBED + gn] = val;
        } else {
          int b = gm >> 11, s = gm & (SEQ - 1);
          int hh = gn >> 6, d = gn & (DH - 1);
          size_t idx = (mode == 0)
              ? ((((size_t)(b * HEADS + hh)) * SEQ + s) * DH + d)
              : ((((size_t)(b * HEADS + hh)) * DH + d) * SEQ + s);
          outh[idx] = (_Float16)val;
        }
      }
    }
  }
}

// ---------------- fused flash attention ----------------
// Q,K: [B,H,S,Dh] f16 ; Vt: [B,H,Dh,S] f16 ; O: [B,S,E] f16
__global__ __launch_bounds__(256)
void flash_attn(const _Float16* __restrict__ Q, const _Float16* __restrict__ K,
                const _Float16* __restrict__ Vt, _Float16* __restrict__ O) {
  __shared__ _Float16 Ks[2][32 * KSTR];     // 32 keys x 64 dims (padded rows)
  __shared__ _Float16 Vs[2][64 * LDSTR];    // 64 dims x 32 keys (padded rows)
  __shared__ _Float16 Ps[8 * 16 * LDSTR];   // per-wave 16x32 prob tile (padded)

  const int t    = threadIdx.x;
  const int wave = t >> 5;
  const int lane = t & 31;
  const int lm   = lane & 15;
  const int hi   = (lane >> 4) & 1;
  const int bh   = blockIdx.y;                      // b*HEADS + h
  const int q0   = blockIdx.x * 128 + wave * 16;
  const int NT   = SEQ / 32;

  const _Float16* Qb = Q  + (size_t)bh * SEQ * DH;
  const _Float16* Kb = K  + (size_t)bh * SEQ * DH;
  const _Float16* Vb = Vt + (size_t)bh * DH * SEQ;
  _Float16* Pw = &Ps[wave * 16 * LDSTR];

  // persistent Q fragments: Dh=64 -> two K=32 steps
  HFrag qf[2];
  #pragma unroll
  for (int dt = 0; dt < 2; ++dt) {
    const _Float16* g = Qb + (size_t)(q0 + lm) * DH + dt * 32 + hi * 8;
    qf[dt].h[0] = *(const v8h*)g;
    qf[dt].h[1] = *(const v8h*)(g + 16);
  }

  v8f zero = {0.f,0.f,0.f,0.f,0.f,0.f,0.f,0.f};
  v8f acc[4];
  #pragma unroll
  for (int nt = 0; nt < 4; ++nt) acc[nt] = zero;
  float m[8], l[8];
  #pragma unroll
  for (int r = 0; r < 8; ++r) { m[r] = -3.0e38f; l[r] = 0.f; }

#if defined(HAVE_TDM)
  if (wave == 0) {   // prologue: stage K/V tile 0 into buffer 0
    tdm_load_2d(Kb, &Ks[0][0], 64, 32, 64, 32, (unsigned long long)DH,  PADF64);
    tdm_load_2d(Vb, &Vs[0][0], 32, 64, 32, 64, (unsigned long long)SEQ, PADF32);
  }
#endif

  for (int ib = 0; ib < NT; ++ib) {
    const int cur = ib & 1;
#if defined(HAVE_TDM)
    if (wave == 0) {
      if (ib + 1 < NT) {   // prefetch next K/V pair
        int kb2 = (ib + 1) * 32;
        tdm_load_2d(Kb + (size_t)kb2 * DH, &Ks[cur ^ 1][0], 64, 32, 64, 32,
                    (unsigned long long)DH, PADF64);
        tdm_load_2d(Vb + kb2, &Vs[cur ^ 1][0], 32, 64, 32, 64,
                    (unsigned long long)SEQ, PADF32);
        __builtin_amdgcn_s_wait_tensorcnt(2);
      } else {
        __builtin_amdgcn_s_wait_tensorcnt(0);
      }
    }
#else
    {
      int kb = ib * 32;
      int row = t >> 3, seg = (t & 7) * 8;   // 32 x 64
      *(v8h*)&Ks[cur][row * KSTR + seg] = *(const v8h*)(Kb + (size_t)(kb + row) * DH + seg);
      int row2 = t >> 2, seg2 = (t & 3) * 8; // 64 x 32
      *(v8h*)&Vs[cur][row2 * LDSTR + seg2] = *(const v8h*)(Vb + (size_t)row2 * SEQ + kb + seg2);
    }
#endif
    __syncthreads();

    // scores for 32 keys = two 16x16 tiles, each over 2 K-steps of Dh
    v8f sc[2];
    #pragma unroll
    for (int hk = 0; hk < 2; ++hk) {
      v8f c = zero;
      #pragma unroll
      for (int dt = 0; dt < 2; ++dt) {
        HFrag kf;  // B matrix: column N = key row (K^T), contiguous along Dh in LDS
        const _Float16* g = &Ks[cur][(hk * 16 + lm) * KSTR + dt * 32 + hi * 8];
        kf.h[0] = *(const v8h*)g;
        kf.h[1] = *(const v8h*)(g + 16);
        c = __builtin_amdgcn_wmma_f32_16x16x32_f16(
            false, qf[dt].v, false, kf.v, (short)0, c, false, false);
      }
      sc[hk] = c * 0.125f;   // 1/sqrt(Dh)
    }

    // online softmax (per accumulator row; 16-lane groups share a row)
    #pragma unroll
    for (int r = 0; r < 8; ++r) {
      float s0 = sc[0][r], s1 = sc[1][r];
      float rm = fmaxf(s0, s1);
      rm = fmaxf(rm, __shfl_xor(rm, 1, 16));
      rm = fmaxf(rm, __shfl_xor(rm, 2, 16));
      rm = fmaxf(rm, __shfl_xor(rm, 4, 16));
      rm = fmaxf(rm, __shfl_xor(rm, 8, 16));
      float mn   = fmaxf(m[r], rm);
      float corr = __expf(m[r] - mn);
      float p0   = __expf(s0 - mn);
      float p1   = __expf(s1 - mn);
      float rs = p0 + p1;
      rs += __shfl_xor(rs, 1, 16);
      rs += __shfl_xor(rs, 2, 16);
      rs += __shfl_xor(rs, 4, 16);
      rs += __shfl_xor(rs, 8, 16);
      l[r] = l[r] * corr + rs;
      m[r] = mn;
      #pragma unroll
      for (int nt = 0; nt < 4; ++nt) acc[nt][r] *= corr;
      int row = r + hi * 8;                         // C layout -> row-major LDS tile
      Pw[row * LDSTR + lm]      = (_Float16)p0;
      Pw[row * LDSTR + 16 + lm] = (_Float16)p1;
    }

    // reload probs as A fragment (wave-private tile: per-wave DS ordering suffices)
    HFrag pf;
    {
      int base = lm * LDSTR + hi * 8;
      pf.h[0] = *(const v8h*)&Pw[base];
      pf.h[1] = *(const v8h*)&Pw[base + 16];
    }
    // P @ V from the shared LDS V tile
    #pragma unroll
    for (int nt = 0; nt < 4; ++nt) {
      HFrag vf;
      const _Float16* g = &Vs[cur][(nt * 16 + lm) * LDSTR + hi * 8];
      vf.h[0] = *(const v8h*)g;
      vf.h[1] = *(const v8h*)(g + 16);
      acc[nt] = __builtin_amdgcn_wmma_f32_16x16x32_f16(
          false, pf.v, false, vf.v, (short)0, acc[nt], false, false);
    }
    __syncthreads();   // readers done before this buffer is DMA-overwritten again
  }

  // normalize and write merged-head output [B,S,E]
  const int b = bh >> 4, h = bh & 15;
  #pragma unroll
  for (int r = 0; r < 8; ++r) {
    float inv = 1.0f / l[r];
    int qrow = q0 + r + hi * 8;
    size_t rowbase = ((size_t)b * SEQ + qrow) * EMBED + h * DH;
    #pragma unroll
    for (int nt = 0; nt < 4; ++nt)
      O[rowbase + nt * 16 + lm] = (_Float16)(acc[nt][r] * inv);
  }
}

extern "C" void kernel_launch(void* const* d_in, const int* in_sizes, int n_in,
                              void* d_out, int out_size, void* d_ws, size_t ws_size,
                              hipStream_t stream) {
  const float* x  = (const float*)d_in[0];
  const float* Wq = (const float*)d_in[1];
  const float* bq = (const float*)d_in[2];
  const float* Wk = (const float*)d_in[3];
  const float* bk = (const float*)d_in[4];
  const float* Wv = (const float*)d_in[5];
  const float* bv = (const float*)d_in[6];
  const float* Wo = (const float*)d_in[7];
  const float* bo = (const float*)d_in[8];
  float* out = (float*)d_out;

  char* ws = (char*)d_ws;
  const size_t MB = 1024ull * 1024ull;
  _Float16* xh  = (_Float16*)(ws);             // 8 MB  [ROWS,EMBED]
  _Float16* Wqt = (_Float16*)(ws + 8  * MB);   // 2 MB  transposed f16 weights
  _Float16* Wkt = (_Float16*)(ws + 10 * MB);
  _Float16* Wvt = (_Float16*)(ws + 12 * MB);
  _Float16* Wot = (_Float16*)(ws + 14 * MB);
  _Float16* Qh  = (_Float16*)(ws + 16 * MB);   // 8 MB  [B,H,S,Dh]
  _Float16* Kh  = (_Float16*)(ws + 24 * MB);   // 8 MB  [B,H,S,Dh]
  _Float16* Vt  = (_Float16*)(ws + 32 * MB);   // 8 MB  [B,H,Dh,S]
  _Float16* Ah  = (_Float16*)(ws + 40 * MB);   // 8 MB  [B,S,E]

  int nx = ROWS * EMBED;
  cvt_f32_f16<<<(nx + 255) / 256, 256, 0, stream>>>(x, xh, nx);
  dim3 gt(EMBED / 32, EMBED / 32);      // (32, 32)
  transpose_cvt_f32_f16<<<gt, 256, 0, stream>>>(Wq, Wqt);
  transpose_cvt_f32_f16<<<gt, 256, 0, stream>>>(Wk, Wkt);
  transpose_cvt_f32_f16<<<gt, 256, 0, stream>>>(Wv, Wvt);
  transpose_cvt_f32_f16<<<gt, 256, 0, stream>>>(Wo, Wot);

  dim3 gg(EMBED / 128, ROWS / 128);     // (8, 32)
  gemm_wmma<<<gg, 256, 0, stream>>>(xh, Wqt, bq, Qh, nullptr, 0);
  gemm_wmma<<<gg, 256, 0, stream>>>(xh, Wkt, bk, Kh, nullptr, 0);
  gemm_wmma<<<gg, 256, 0, stream>>>(xh, Wvt, bv, Vt, nullptr, 1);

  dim3 ga(SEQ / 128, BATCH * HEADS);    // (16, 32)
  flash_attn<<<ga, 256, 0, stream>>>(Qh, Kh, Vt, Ah);

  gemm_wmma<<<gg, 256, 0, stream>>>(Ah, Wot, bo, nullptr, out, 2);
}
